// MistralAttention_42631845380113
// MI455X (gfx1250) — matrix operand order repaired and checked by reading, not compiled, not run on hardware
//
#include <hip/hip_runtime.h>
#include <hip/hip_bf16.h>
#include <math.h>

typedef __bf16 bf16;
typedef __attribute__((ext_vector_type(16))) __bf16 v16bf;
typedef __attribute__((ext_vector_type(8)))  float  v8f;
typedef __attribute__((ext_vector_type(4)))  unsigned int uint4v;
typedef __attribute__((ext_vector_type(4)))  unsigned int u32x4;
typedef __attribute__((ext_vector_type(8)))  int i32x8;
typedef __attribute__((ext_vector_type(4)))  int i32x4;

#define HIDDEN 4096
#define SEQ    2048
#define HEAD   128
#define NQH    32
#define NKVH   8
#define KVDIM  1024

// ---------------- WMMA fragment helpers (CDNA5 16x16x32 bf16) ----------------
static __device__ __forceinline__ v16bf lds_fragA(const bf16* base, int rowBase, int ldr, int kBase) {
  const int lane = threadIdx.x & 31;
  const int r  = rowBase + (lane & 15);
  const int ks = (lane >> 4) * 8;
  const bf16* p = base + r * ldr + kBase + ks;
  union { uint4v u[2]; v16bf v; } f;
  f.u[0] = *(const uint4v*)(p);        // K = kBase+ks .. +7
  f.u[1] = *(const uint4v*)(p + 16);   // K = kBase+16+ks .. +7
  return f.v;
}

static __device__ __forceinline__ v16bf lds_fragB(const bf16* base, int colBase, int ldr, int kBase) {
  const int lane = threadIdx.x & 31;
  const int n  = colBase + (lane & 15);
  const int ks = (lane >> 4) * 8;
  const bf16* p = base + n * ldr + kBase + ks;
  union { uint4v u[2]; v16bf v; } f;
  f.u[0] = *(const uint4v*)(p);
  f.u[1] = *(const uint4v*)(p + 16);
  return f.v;
}

static __device__ __forceinline__ v8f wmma_bf16(v16bf a, v16bf b, v8f c) {
  return __builtin_amdgcn_wmma_f32_16x16x32_bf16(false, a, false, b, (short)0, c, false, false);
}

// ---------------- TDM: async 2-D bf16 tile load, global -> LDS ----------------
// D# per ISA 8.3/8.4: count=1, type=2 ("image"), data_size=1 (2 bytes),
// pad_enable with pad of 4 DWORDs (8 bf16) after every 16 DWORDs (32 bf16 row)
// -> reproduces the LD=40 bf16 LDS row stride in hardware.
static __device__ __forceinline__ void tdm_load_tile_2d(
    unsigned lds_off_bytes, const void* gaddr,
    unsigned tensor_d0, unsigned tensor_d1,
    unsigned tile_d0, unsigned tile_d1,
    unsigned long long stride0_elems) {
  unsigned long long ga = (unsigned long long)(uintptr_t)gaddr;
  u32x4 g0;
  g0[0] = 1u;                                     // count=1, user mode
  g0[1] = lds_off_bytes;                          // lds_addr
  g0[2] = (unsigned)(ga & 0xFFFFFFFFu);           // global_addr[31:0]
  g0[3] = (unsigned)((ga >> 32) & 0x1FFFFFFu) | (2u << 30);  // [56:32] | type=2
  i32x8 g1;
  g1[0] = (int)((1u << 16)   // data_size = 2 bytes
              | (1u << 20)   // pad_enable
              | (3u << 22)   // pad_interval: every 16 DWORDs (64B row)
              | (3u << 25)); // pad_amount: 4 DWORDs (16B)
  g1[1] = (int)((tensor_d0 & 0xFFFFu) << 16);                              // dim0[15:0]
  g1[2] = (int)(((tensor_d0 >> 16) & 0xFFFFu) | ((tensor_d1 & 0xFFFFu) << 16));
  g1[3] = (int)(((tensor_d1 >> 16) & 0xFFFFu) | (tile_d0 << 16));          // tile_dim0
  g1[4] = (int)(tile_d1 & 0xFFFFu);                                        // tile_dim1, tile_dim2=0
  g1[5] = (int)(stride0_elems & 0xFFFFFFFFull);                            // dim0 stride lo
  g1[6] = (int)((stride0_elems >> 32) & 0xFFFFull);                        // stride hi, dim1 stride=0
  g1[7] = 0;
  i32x4 z4 = {0, 0, 0, 0};
#if defined(__clang_major__) && (__clang_major__ >= 23)
  i32x8 z8 = {0, 0, 0, 0, 0, 0, 0, 0};
  __builtin_amdgcn_tensor_load_to_lds(g0, g1, z4, z4, z8, 0);
#else
  __builtin_amdgcn_tensor_load_to_lds(g0, g1, z4, z4, 0);
#endif
}

// ---------------- conversion kernels ----------------
__global__ __launch_bounds__(256)
void cvt_f32_bf16(const float* __restrict__ x, bf16* __restrict__ y, int n) {
  int i = blockIdx.x * 256 + threadIdx.x;
  if (i < n) y[i] = (bf16)x[i];
}

// W [R][C] f32 row-major  ->  Wt [C][R] bf16 row-major
__global__ __launch_bounds__(256)
void transpose_cvt(const float* __restrict__ W, bf16* __restrict__ Wt, int R, int C) {
  __shared__ float tile[32][33];
  const int tx = threadIdx.x & 31, ty = threadIdx.x >> 5;
  const int c0 = blockIdx.x * 32, r0 = blockIdx.y * 32;
#pragma unroll
  for (int i = 0; i < 32; i += 8)
    tile[ty + i][tx] = W[(size_t)(r0 + ty + i) * C + (c0 + tx)];
  __syncthreads();
#pragma unroll
  for (int i = 0; i < 32; i += 8)
    Wt[(size_t)(c0 + ty + i) * R + (r0 + tx)] = (bf16)tile[tx][ty + i];
}

// ------- WMMA GEMM with TDM double-buffered LDS: C[M,N] = A[M,K] * Bt[N,K]^T -
template<bool STORE_F32>
__global__ __launch_bounds__(256)
void gemm_bf16(const bf16* __restrict__ A, const bf16* __restrict__ Bt,
               float* __restrict__ Cf, bf16* __restrict__ Cb,
               int M, int N, int K) {
  constexpr int LD = 40;                         // 32 + 8 bf16 pad (TDM-generated)
  constexpr unsigned TILE_BYTES = 128u * LD * 2; // 10240 B per tile
  __shared__ __align__(16) bf16 smem[2 * 2 * 128 * LD];  // [buf][{A,B}]
  const int tid  = threadIdx.x;
  const int wave = tid >> 5, lane = tid & 31;
  const int wm = wave >> 1, wn = wave & 1;       // 4x2 wave grid
  const size_t bm = (size_t)blockIdx.x * 128, bn = (size_t)blockIdx.y * 128;
  const v8f vz = {0.f,0.f,0.f,0.f,0.f,0.f,0.f,0.f};
  v8f acc[2][4];
#pragma unroll
  for (int i = 0; i < 2; ++i)
#pragma unroll
    for (int j = 0; j < 4; ++j) acc[i][j] = vz;

  // wave 0 issues a pair of TDM tile loads (A 128x32, B 128x32) for buffer b
  auto issue = [&](int b, int k0) {
    unsigned aoff = (unsigned)b * 2u * TILE_BYTES;
    tdm_load_tile_2d(aoff, A + bm * (size_t)K + k0,
                     (unsigned)K, (unsigned)M, 32u, 128u, (unsigned long long)K);
    tdm_load_tile_2d(aoff + TILE_BYTES, Bt + bn * (size_t)K + k0,
                     (unsigned)K, (unsigned)N, 32u, 128u, (unsigned long long)K);
  };

  if (tid < 32) issue(0, 0);

  int buf = 0;
  for (int k0 = 0; k0 < K; k0 += 32, buf ^= 1) {
    const bool more = (k0 + 32) < K;
    if (tid < 32) {
      if (more) issue(buf ^ 1, k0 + 32);          // prefetch next tile via TDM
      if (more) __builtin_amdgcn_s_wait_tensorcnt(2);  // current pair done
      else      __builtin_amdgcn_s_wait_tensorcnt(0);
    }
    __syncthreads();

    const bf16* As = smem + (size_t)buf * 2u * 128u * LD;
    const bf16* Bs = As + 128 * LD;
    v16bf af[2], bfr[4];
    af[0] = lds_fragA(As, wm * 32,      LD, 0);
    af[1] = lds_fragA(As, wm * 32 + 16, LD, 0);
#pragma unroll
    for (int nf = 0; nf < 4; ++nf) bfr[nf] = lds_fragB(Bs, wn * 64 + nf * 16, LD, 0);
#pragma unroll
    for (int mf = 0; mf < 2; ++mf)
#pragma unroll
      for (int nf = 0; nf < 4; ++nf)
        acc[mf][nf] = wmma_bf16(af[mf], bfr[nf], acc[mf][nf]);
    __syncthreads();                              // all reads done before re-issue
  }

  const int rlo = (lane >> 4) * 8, nin = lane & 15;
#pragma unroll
  for (int mf = 0; mf < 2; ++mf)
#pragma unroll
    for (int nf = 0; nf < 4; ++nf)
#pragma unroll
      for (int r = 0; r < 8; ++r) {
        size_t row = bm + wm * 32 + mf * 16 + r + rlo;
        size_t col = bn + wn * 64 + nf * 16 + nin;
        float v = acc[mf][nf][r];
        if (STORE_F32) Cf[row * (size_t)N + col] = v;
        else           Cb[row * (size_t)N + col] = (bf16)v;
      }
}

// ---------------- RoPE (in place, bf16) ----------------
__global__ __launch_bounds__(256)
void rope_bf16(bf16* __restrict__ X, const int* __restrict__ pos, int heads) {
  int idx = blockIdx.x * 256 + threadIdx.x;   // one thread per (s, h, d<64)
  int d = idx & 63;
  int h = (idx >> 6) % heads;
  int s = idx / (64 * heads);
  float p = (float)pos[s];
  float ang = p * __powf(10000.f, -(float)d * (1.f / 64.f));
  float sn, cs;
  __sincosf(ang, &sn, &cs);
  bf16* q = X + ((size_t)s * heads + h) * HEAD;
  float x1 = (float)q[d], x2 = (float)q[d + 64];
  q[d]      = (bf16)(x1 * cs - x2 * sn);
  q[d + 64] = (bf16)(x2 * cs + x1 * sn);
}

// ---------------- flash attention: 4 waves, 64 q-rows, 32-key blocks ----------
__global__ __launch_bounds__(128)
void flash_attn(const bf16* __restrict__ Q, const bf16* __restrict__ Kb,
                const bf16* __restrict__ Vb, bf16* __restrict__ O) {
  constexpr int LQ = 136, LV = 40, LS = 36, LP = 40;
  __shared__ bf16  Qs [64 * LQ];     // 64 q-rows x 128
  __shared__ bf16  Ks [32 * LQ];     // 32 keys   x 128
  __shared__ bf16  Vts[128 * LV];    // V^T: 128 d-rows x 32 keys
  __shared__ float Sf [64 * LS];     // raw scores
  __shared__ bf16  Ps [64 * LP];     // exp probs (bf16)
  __shared__ float m_i[64], l_i[64], srow[64];

  const int tid  = threadIdx.x;
  const int wave = tid >> 5, lane = tid & 31;
  const int h   = blockIdx.x;
  const int q0  = blockIdx.y * 64;
  const int kvh = h >> 2;                          // GQA: 4 q-heads per kv head
  const int rlo = (lane >> 4) * 8, nin = lane & 15;
  const float scale = 0.08838834764831845f;        // 1/sqrt(128)
  const v8f vz = {0.f,0.f,0.f,0.f,0.f,0.f,0.f,0.f};

#pragma unroll
  for (int i = 0; i < 8; ++i) {
    int chunk = tid + i * 128;
    int r = chunk >> 4, c = (chunk & 15) * 8;
    *(uint4v*)(Qs + r * LQ + c) =
        *(const uint4v*)(Q + ((size_t)(q0 + r) * NQH + h) * HEAD + c);
  }
  if (tid < 64) { m_i[tid] = -1e30f; l_i[tid] = 0.f; }
  v8f oA[8];
#pragma unroll
  for (int nf = 0; nf < 8; ++nf) oA[nf] = vz;

  for (int k0 = 0; k0 < q0 + 64; k0 += 32) {
#pragma unroll
    for (int i = 0; i < 4; ++i) {
      int chunk = tid + i * 128;
      int r = chunk >> 4, c = (chunk & 15) * 8;
      *(uint4v*)(Ks + r * LQ + c) =
          *(const uint4v*)(Kb + ((size_t)(k0 + r) * NKVH + kvh) * HEAD + c);
    }
    {
      int kk = tid & 31, dbase = (tid >> 5) * 32;
      union { uint4v u[4]; bf16 e[32]; } vr;
      const bf16* src = Vb + ((size_t)(k0 + kk) * NKVH + kvh) * HEAD + dbase;
#pragma unroll
      for (int j = 0; j < 4; ++j) vr.u[j] = *(const uint4v*)(src + j * 8);
#pragma unroll
      for (int j = 0; j < 32; ++j) Vts[(dbase + j) * LV + kk] = vr.e[j];
    }
    __syncthreads();

    v8f sA[2]; sA[0] = vz; sA[1] = vz;
#pragma unroll
    for (int kk = 0; kk < 4; ++kk) {
      v16bf a  = lds_fragA(Qs, 16 * wave, LQ, 32 * kk);
      v16bf b0 = lds_fragB(Ks, 0,  LQ, 32 * kk);
      v16bf b1 = lds_fragB(Ks, 16, LQ, 32 * kk);
      sA[0] = wmma_bf16(a, b0, sA[0]);
      sA[1] = wmma_bf16(a, b1, sA[1]);
    }
#pragma unroll
    for (int nf = 0; nf < 2; ++nf)
#pragma unroll
      for (int r = 0; r < 8; ++r) {
        int ml = 16 * wave + r + rlo;
        int kg = k0 + nf * 16 + nin;
        float v = sA[nf][r] * scale;
        if (kg > q0 + ml) v = -1e9f;               // causal mask (reference uses -1e9)
        Sf[ml * LS + nf * 16 + nin] = v;
      }
    __syncthreads();

    if (tid < 64) {
      if (k0 <= q0 + tid) {
        float mo = m_i[tid], mx = mo;
        for (int j = 0; j < 32; ++j) mx = fmaxf(mx, Sf[tid * LS + j]);
        float corr = __expf(mo - mx);
        float sum = 0.f;
        for (int j = 0; j < 32; ++j) {
          float e = __expf(Sf[tid * LS + j] - mx);
          sum += e;
          Ps[tid * LP + j] = (bf16)e;
        }
        m_i[tid] = mx;
        l_i[tid] = l_i[tid] * corr + sum;
        srow[tid] = corr;
      } else {
        for (int j = 0; j < 32; ++j) Ps[tid * LP + j] = (bf16)0.f;
        srow[tid] = 1.f;
      }
    }
    __syncthreads();

#pragma unroll
    for (int r = 0; r < 8; ++r) {
      float s = srow[16 * wave + r + rlo];
#pragma unroll
      for (int nf = 0; nf < 8; ++nf) oA[nf][r] *= s;
    }
    v16bf p = lds_fragA(Ps, 16 * wave, LP, 0);
#pragma unroll
    for (int nf = 0; nf < 8; ++nf) {
      v16bf b = lds_fragB(Vts, nf * 16, LV, 0);
      oA[nf] = wmma_bf16(p, b, oA[nf]);
    }
    __syncthreads();
  }

#pragma unroll
  for (int r = 0; r < 8; ++r) {
    int ml = 16 * wave + r + rlo;
    float inv = 1.f / l_i[ml];
#pragma unroll
    for (int nf = 0; nf < 8; ++nf)
      O[((size_t)(q0 + ml) * NQH + h) * HEAD + nf * 16 + nin] = (bf16)(oA[nf][r] * inv);
  }
}

// ---------------- launcher ----------------
extern "C" void kernel_launch(void* const* d_in, const int* in_sizes, int n_in,
                              void* d_out, int out_size, void* d_ws, size_t ws_size,
                              hipStream_t stream) {
  (void)in_sizes; (void)n_in; (void)out_size; (void)ws_size;
  const float* hs = (const float*)d_in[0];
  const float* wq = (const float*)d_in[1];
  const float* wk = (const float*)d_in[2];
  const float* wv = (const float*)d_in[3];
  const float* wo = (const float*)d_in[4];
  const int*  pos = (const int*)d_in[6];
  float* out = (float*)d_out;

  char* ws = (char*)d_ws;
  size_t off = 0;
  auto alloc = [&](size_t bytes) -> void* {
    void* p = ws + off; off += (bytes + 255) & ~(size_t)255; return p;
  };
  bf16* Xb  = (bf16*)alloc((size_t)SEQ * HIDDEN * 2);
  bf16* Wqt = (bf16*)alloc((size_t)HIDDEN * HIDDEN * 2);
  bf16* Wkt = (bf16*)alloc((size_t)KVDIM * HIDDEN * 2);
  bf16* Wvt = (bf16*)alloc((size_t)KVDIM * HIDDEN * 2);
  bf16* Wot = (bf16*)alloc((size_t)HIDDEN * HIDDEN * 2);
  bf16* Qb  = (bf16*)alloc((size_t)SEQ * HIDDEN * 2);
  bf16* Kbp = (bf16*)alloc((size_t)SEQ * KVDIM * 2);
  bf16* Vbp = (bf16*)alloc((size_t)SEQ * KVDIM * 2);
  bf16* Ob  = (bf16*)alloc((size_t)SEQ * HIDDEN * 2);

  cvt_f32_bf16<<<SEQ * HIDDEN / 256, 256, 0, stream>>>(hs, Xb, SEQ * HIDDEN);
  transpose_cvt<<<dim3(HIDDEN / 32, HIDDEN / 32), 256, 0, stream>>>(wq, Wqt, HIDDEN, HIDDEN);
  transpose_cvt<<<dim3(KVDIM  / 32, HIDDEN / 32), 256, 0, stream>>>(wk, Wkt, HIDDEN, KVDIM);
  transpose_cvt<<<dim3(KVDIM  / 32, HIDDEN / 32), 256, 0, stream>>>(wv, Wvt, HIDDEN, KVDIM);
  transpose_cvt<<<dim3(HIDDEN / 32, HIDDEN / 32), 256, 0, stream>>>(wo, Wot, HIDDEN, HIDDEN);

  gemm_bf16<false><<<dim3(SEQ / 128, HIDDEN / 128), 256, 0, stream>>>(Xb, Wqt, nullptr, Qb,  SEQ, HIDDEN, HIDDEN);
  gemm_bf16<false><<<dim3(SEQ / 128, KVDIM  / 128), 256, 0, stream>>>(Xb, Wkt, nullptr, Kbp, SEQ, KVDIM,  HIDDEN);
  gemm_bf16<false><<<dim3(SEQ / 128, KVDIM  / 128), 256, 0, stream>>>(Xb, Wvt, nullptr, Vbp, SEQ, KVDIM,  HIDDEN);

  rope_bf16<<<SEQ * NQH  * 64 / 256, 256, 0, stream>>>(Qb,  pos, NQH);
  rope_bf16<<<SEQ * NKVH * 64 / 256, 256, 0, stream>>>(Kbp, pos, NKVH);

  flash_attn<<<dim3(NQH, SEQ / 64), 128, 0, stream>>>(Qb, Kbp, Vbp, Ob);

  gemm_bf16<true><<<dim3(SEQ / 128, HIDDEN / 128), 256, 0, stream>>>(Ob, Wot, out, nullptr, SEQ, HIDDEN, HIDDEN);
}
